// IPAUpdateLayer_40527311405662
// MI455X (gfx1250) — matrix operand order repaired and checked by reading, not compiled, not run on hardware
//
#include <hip/hip_runtime.h>
#include <hip/hip_bf16.h>

typedef __attribute__((ext_vector_type(16))) _Float16 v16h;
typedef __attribute__((ext_vector_type(8)))  _Float16 v8h;
typedef __attribute__((ext_vector_type(8)))  float    v8f;
typedef __attribute__((ext_vector_type(4)))  unsigned int u32x4;
typedef __attribute__((ext_vector_type(8)))  unsigned int u32x8;

#define NRES 384
#define CS   384
#define CZ   128
#define CH   16
#define PQK  8
#define PV   12
#define NH   16
#define HID  512
#define CATW 3072
#define INFV 100000.0f

#if __has_builtin(__builtin_amdgcn_s_wait_asynccnt)
#define AWAIT0() __builtin_amdgcn_s_wait_asynccnt(0)
#else
#define AWAIT0() asm volatile("s_wait_asynccnt 0" ::: "memory")
#endif

#if __has_builtin(__builtin_amdgcn_s_wait_tensorcnt)
#define TWAIT0() __builtin_amdgcn_s_wait_tensorcnt(0)
#else
#define TWAIT0() asm volatile("s_wait_tensorcnt 0" ::: "memory")
#endif

__device__ __forceinline__ v16h cat8(v8h lo, v8h hi) {
  return __builtin_shufflevector(lo, hi, 0,1,2,3,4,5,6,7,8,9,10,11,12,13,14,15);
}

// ---------------------------------------------------------------------------
// Generic batched GEMM: C[b,m,n] = act( sum_k A[b,m,k]*W[b,k,n] + bias[n] )
// fp32 in memory, f16 in LDS, f32 WMMA accumulation (v_wmma_f32_16x16x32_f16).
// Block = 256 threads = 8 waves; tile 64(M) x 32(N); waves: 4 in M x 2 in N.
// ---------------------------------------------------------------------------
template<bool RELU>
__global__ __launch_bounds__(256)
void gemm_wmma(const float* __restrict__ A, const float* __restrict__ W,
               const float* __restrict__ bias, float* __restrict__ C,
               int M, int K, int Nc,
               long ldA, long ldW, long ldC,
               long sA, long sW, long sC)
{
  __shared__ __align__(16) _Float16 As[64][40];   // [m][k]  pitch 80B
  __shared__ __align__(16) _Float16 Bs[32][40];   // [n][k]  pitch 80B

  const int tid  = threadIdx.x;
  const int wave = tid >> 5, lane = tid & 31;
  const int wm = wave & 3, wn = wave >> 2;
  const int ml = lane & 15, half = lane >> 4;
  const long m0 = (long)blockIdx.x * 64;
  const long n0 = (long)blockIdx.y * 32;
  A += (long)blockIdx.z * sA;
  W += (long)blockIdx.z * sW;
  C += (long)blockIdx.z * sC;

  v8f acc = {};
  for (int k0 = 0; k0 < K; k0 += 32) {
    { // A tile 64x32: 8 elems/thread
      int r  = tid >> 2;
      int kb = (tid & 3) * 8;
      long row = m0 + r;
      #pragma unroll
      for (int e = 0; e < 8; ++e) {
        float v = 0.f;
        if (row < M) v = A[row * ldA + (k0 + kb + e)];
        As[r][kb + e] = (_Float16)v;
      }
    }
    { // W tile 32x32 -> Bs[n][k] (transposed store): 4 elems/thread
      #pragma unroll
      for (int e = 0; e < 4; ++e) {
        int li = e * 256 + tid;
        int kk = li >> 5;
        int nn = li & 31;
        float v = 0.f;
        if (n0 + nn < Nc) v = W[(long)(k0 + kk) * ldW + (n0 + nn)];
        Bs[nn][kk] = (_Float16)v;
      }
    }
    if (k0 + 32 < K) __builtin_prefetch(&W[(long)(k0 + 32) * ldW + n0], 0, 1);
    __syncthreads();
    v8h alo = *(const v8h*)&As[wm*16 + ml][half*8];
    v8h ahi = *(const v8h*)&As[wm*16 + ml][16 + half*8];
    v8h blo = *(const v8h*)&Bs[wn*16 + ml][half*8];
    v8h bhi = *(const v8h*)&Bs[wn*16 + ml][16 + half*8];
    v16h af = cat8(alo, ahi);
    v16h bf = cat8(blo, bhi);
    acc = __builtin_amdgcn_wmma_f32_16x16x32_f16(false, af, false, bf, (short)0, acc, false, false);
    __syncthreads();
  }
  long col = n0 + wn*16 + ml;
  if (col < Nc) {
    float bv = bias ? bias[col] : 0.f;
    long rbase = m0 + wm*16 + half*8;
    #pragma unroll
    for (int r = 0; r < 8; ++r) {
      long row = rbase + r;
      if (row < M) {
        float v = acc[r] + bv;
        if (RELU) v = v > 0.f ? v : 0.f;
        C[row * ldC + col] = v;
      }
    }
  }
}

// ---------------------------------------------------------------------------
// One-time f32 -> transposed f16 weight conversion: out[n][k] = in[k][n]
// ---------------------------------------------------------------------------
__global__ void convert_transpose_f16(const float* __restrict__ in, _Float16* __restrict__ out,
                                      int K, int Ncols)
{
  int idx = blockIdx.x * 256 + threadIdx.x;
  if (idx >= K * Ncols) return;
  int n = idx / K, k = idx % K;
  out[idx] = (_Float16)in[(long)k * Ncols + n];
}

// ---------------------------------------------------------------------------
// Rigid transform: out = R_n @ p + t_n    (one thread per point)
// ---------------------------------------------------------------------------
__global__ void transform_points(const float* __restrict__ raw, const float* __restrict__ rots,
                                 const float* __restrict__ trans, float* __restrict__ out, int ppn)
{
  int idx = blockIdx.x * blockDim.x + threadIdx.x;
  if (idx >= NRES * ppn) return;
  int n = idx / ppn;
  const float* v = raw + (long)idx * 3;
  const float* R = rots + n * 9;
  const float* t = trans + n * 3;
  float x = v[0], y = v[1], z = v[2];
  out[idx*3+0] = R[0]*x + R[1]*y + R[2]*z + t[0];
  out[idx*3+1] = R[3]*x + R[4]*y + R[5]*z + t[1];
  out[idx*3+2] = R[6]*x + R[7]*y + R[8]*z + t[2];
}

// ---------------------------------------------------------------------------
// Build packed score vectors (fold scales + point-norm terms into rank-1 form)
// u = [s1*q(16), cw*qp(24), 1, -cw/2*|qp|^2, 0pad] ; v = [k, kp, -cw/2*|kp|^2, 1, 0pad]
// so u.v = s1*qk - 0.5*cw*sum_p |qp-kp|^2.  Also builds Vcat = [v(16), v_pts(36), pad].
// ---------------------------------------------------------------------------
__global__ void build_packs(const float* __restrict__ q_raw, const float* __restrict__ kv_raw,
                            const float* __restrict__ qpts, const float* __restrict__ kvpts,
                            const float* __restrict__ head_w,
                            float* __restrict__ Qp, float* __restrict__ Kt, float* __restrict__ Vc)
{
  int idx = blockIdx.x * blockDim.x + threadIdx.x;
  if (idx >= NH * NRES) return;
  int h = idx / NRES, n = idx % NRES;
  float hw = head_w[h];
  hw = (hw > 20.f) ? hw : log1pf(__expf(hw));          // softplus
  const float s_qk = 0.14433756729740643f;             // sqrt(1/(3*16))
  const float wc   = 0.09622504486493764f;             // sqrt(1/108)
  float cw = wc * hw;

  float* u = Qp + ((long)h * NRES + n) * 64;
  const float* q  = q_raw + (long)n * (NH*CH) + h * CH;
  const float* qp = qpts + ((long)n * NH * PQK + h * PQK) * 3;
  float qn2 = 0.f;
  for (int c = 0; c < 16; ++c) u[c] = q[c] * s_qk;
  for (int d = 0; d < 24; ++d) { float x = qp[d]; u[16+d] = cw * x; qn2 += x*x; }
  u[40] = 1.0f;
  u[41] = -0.5f * cw * qn2;
  for (int d = 42; d < 64; ++d) u[d] = 0.f;

  float* vt = Kt + (long)h * 64 * NRES + n;            // [h][d][j], stride NRES per d
  const float* kv = kv_raw + (long)n * (NH*2*CH) + h * (2*CH);
  const float* kp = kvpts + ((long)n * NH * (PQK+PV) + h * (PQK+PV)) * 3;
  float kn2 = 0.f;
  for (int c = 0; c < 16; ++c) vt[(long)c * NRES] = kv[c];
  for (int d = 0; d < 24; ++d) { float x = kp[d]; vt[(long)(16+d)*NRES] = x; kn2 += x*x; }
  vt[(long)40*NRES] = -0.5f * cw * kn2;
  vt[(long)41*NRES] = 1.0f;
  for (int d = 42; d < 64; ++d) vt[(long)d*NRES] = 0.f;

  float* vc = Vc + ((long)h * NRES + n) * 64;          // [v(16), v_pts(36), pad]
  for (int c = 0; c < 16; ++c) vc[c] = kv[16 + c];
  const float* vp = kvpts + ((long)n * NH * (PQK+PV) + h * (PQK+PV) + PQK) * 3;
  for (int d = 0; d < 36; ++d) vc[16+d] = vp[d];
  for (int d = 52; d < 64; ++d) vc[d] = 0.f;
}

// ---------------------------------------------------------------------------
// Softmax rows (in-place on att[h][i][:]); adds z-bias and mask term.
// ---------------------------------------------------------------------------
__global__ __launch_bounds__(128)
void softmax_rows(float* __restrict__ att, const float* __restrict__ bz,
                  const float* __restrict__ mask)
{
  int h = blockIdx.x / NRES, i = blockIdx.x % NRES;
  float* row = att + (long)blockIdx.x * NRES;
  const float sb = 0.5773502691896258f;                // sqrt(1/3)
  float mi = mask[i];
  __shared__ float red[8];
  int lane = threadIdx.x & 31, wv = threadIdx.x >> 5;
  float vals[3];
  float lmax = -3.0e38f;
  #pragma unroll
  for (int t = 0; t < 3; ++t) {
    int j = threadIdx.x + t*128;
    float x = row[j] + sb * bz[((long)i*NRES + j)*NH + h] + (mi*mask[j] - 1.f) * INFV;
    vals[t] = x;
    lmax = fmaxf(lmax, x);
  }
  for (int o = 16; o; o >>= 1) lmax = fmaxf(lmax, __shfl_xor(lmax, o, 32));
  if (lane == 0) red[wv] = lmax;
  __syncthreads();
  lmax = fmaxf(fmaxf(red[0], red[1]), fmaxf(red[2], red[3]));
  float lsum = 0.f;
  #pragma unroll
  for (int t = 0; t < 3; ++t) { vals[t] = __expf(vals[t] - lmax); lsum += vals[t]; }
  for (int o = 16; o; o >>= 1) lsum += __shfl_xor(lsum, o, 32);
  __syncthreads();
  if (lane == 0) red[4 + wv] = lsum;
  __syncthreads();
  lsum = red[4] + red[5] + red[6] + red[7];
  float inv = 1.f / lsum;
  #pragma unroll
  for (int t = 0; t < 3; ++t) row[threadIdx.x + t*128] = vals[t] * inv;
}

// ---------------------------------------------------------------------------
// Assemble cat = [o(256), o_pt_local(576), |o_pt|(192), o_pair(2048)], with
// inverse rigid transform applied to o_pt.
// ---------------------------------------------------------------------------
__global__ __launch_bounds__(256)
void finalize_cat(const float* __restrict__ Outb, const float* __restrict__ Opair,
                  const float* __restrict__ rots, const float* __restrict__ trans,
                  float* __restrict__ cat)
{
  int i = blockIdx.x;
  const float* R = rots + i*9;
  const float* t = trans + i*3;
  float* crow = cat + (long)i * CATW;
  for (int idx = threadIdx.x; idx < NH*CH; idx += 256) {
    int h = idx >> 4, c = idx & 15;
    crow[idx] = Outb[((long)h*NRES + i)*64 + c];
  }
  for (int idx = threadIdx.x; idx < NH*PV; idx += 256) {
    int h = idx / PV, p = idx % PV;
    const float* og = Outb + ((long)h*NRES + i)*64 + 16 + p*3;
    float x = og[0]-t[0], y = og[1]-t[1], z = og[2]-t[2];
    float lx = R[0]*x + R[3]*y + R[6]*z;               // R^T
    float ly = R[1]*x + R[4]*y + R[7]*z;
    float lz = R[2]*x + R[5]*y + R[8]*z;
    crow[256 + idx*3 + 0] = lx;
    crow[256 + idx*3 + 1] = ly;
    crow[256 + idx*3 + 2] = lz;
    crow[832 + idx] = sqrtf(lx*lx + ly*ly + lz*lz + 1e-8f);
  }
  for (int idx = threadIdx.x; idx < NH*CZ; idx += 256)
    crow[1024 + idx] = Opair[(long)i*NH*CZ + idx];
}

// ---------------------------------------------------------------------------
// Row LayerNorm. MODE 0: x = X + D*mask_i, out = LN(x).
//                MODE 1: x = X + D, out = LN(x)*mask_i.
// ---------------------------------------------------------------------------
template<int MODE>
__global__ __launch_bounds__(128)
void layernorm_row(const float* __restrict__ X, const float* __restrict__ Dlt,
                   const float* __restrict__ mask, const float* __restrict__ g,
                   const float* __restrict__ bta, float* __restrict__ out, int D)
{
  int i = blockIdx.x;
  __shared__ float red[8];
  float mi = mask[i];
  float s = 0.f, s2 = 0.f;
  for (int c = threadIdx.x; c < D; c += 128) {
    float d = Dlt[(long)i*D + c];
    float x = X[(long)i*D + c] + (MODE == 0 ? d * mi : d);
    s += x; s2 += x*x;
  }
  int lane = threadIdx.x & 31, wv = threadIdx.x >> 5;
  for (int o = 16; o; o >>= 1) { s += __shfl_xor(s, o, 32); s2 += __shfl_xor(s2, o, 32); }
  if (lane == 0) { red[wv] = s; red[4+wv] = s2; }
  __syncthreads();
  s  = red[0]+red[1]+red[2]+red[3];
  s2 = red[4]+red[5]+red[6]+red[7];
  float mean = s / D;
  float var  = s2 / D - mean*mean;
  float rs = rsqrtf(var + 1e-5f);
  for (int c = threadIdx.x; c < D; c += 128) {
    float d = Dlt[(long)i*D + c];
    float x = X[(long)i*D + c] + (MODE == 0 ? d * mi : d);
    float y = (x - mean) * rs * g[c] + bta[c];
    if (MODE == 1) y *= mi;
    out[(long)i*D + c] = y;
  }
}

// ---------------------------------------------------------------------------
// Fused edge transition with CDNA5 TDM + async weight staging.
// Per 16-pair tile: e=[z|nb_i|nb_j](512) -> relu(e@W1+b1) -> relu(h@W2+b2)
// -> LN((h+e)@Wf+bf).  e/h tiles live in LDS (never hit HBM).  W1/W2 slabs
// are moved by the Tensor Data Mover (tensor_load_to_lds, TENSORcnt), with
// the TDM pad feature producing the 80B bank-staggered LDS pitch; Wf slabs
// use per-lane GLOBAL_LOAD_ASYNC_TO_LDS_B128 (ASYNCcnt).  Double-buffered.
// ---------------------------------------------------------------------------
#define EP 520     // e/h tile pitch (f16), rows 16B aligned, bank-staggered
#define WP 40      // weight-slab pitch (f16), 80B rows

// Per-lane async copy of one 32-k slab of Wt[rows][Kfull] into Ws[rows][WP].
__device__ __forceinline__ void async_slab(const _Float16* __restrict__ Wt,
                                           int rows, int Kfull, int k0,
                                           _Float16 (*Ws)[WP], int tid)
{
  int chunks = rows * 4;                       // 4 x 16B chunks per 64B row
  for (int c = tid; c < chunks; c += 256) {
    int n = c >> 2, part = c & 3;
    unsigned int lds  = (unsigned int)(size_t)&Ws[n][part * 8];
    unsigned int voff = (unsigned int)((n * Kfull + k0 + part * 8) * 2);
    asm volatile("global_load_async_to_lds_b128 %0, %1, %2 offset:0"
                 :: "v"(lds), "v"(voff), "s"(Wt)
                 : "memory");
  }
}

// One TDM descriptor moves the whole 2D slab (rows x 64B) with LDS padding:
// pad_interval=3 (every 16 DWORDs) + pad_amount=3 (4 DWORDs) => 80B pitch.
// D# per ISA 8.3/8.4: group0 = {count|flags, lds_addr, global_addr, type},
// group1 = {mask/size/pad, dims, tile dims, strides}.  Issued by one wave.
__device__ __forceinline__ void tdm_slab(const _Float16* __restrict__ Wt,
                                         int rows, int Kfull, int k0,
                                         _Float16 (*Ws)[WP])
{
  unsigned long long ga = (unsigned long long)(size_t)Wt + (unsigned long long)k0 * 2ull;
  u32x4 g0;
  g0.x = 1u;                                            // count=1 (valid), user mode
  g0.y = (unsigned int)(size_t)&Ws[0][0];               // lds_addr (LDS aperture low 32b)
  g0.z = (unsigned int)(ga & 0xFFFFFFFFu);              // global_addr[31:0] (tile start)
  g0.w = (unsigned int)((ga >> 32) & 0x1FFFFFFu) | (2u << 30);  // addr[56:32] | type=2
  u32x8 g1;
  g1.s0 = (1u << 16) | (1u << 20) | (3u << 22) | (3u << 25); // data_size=2B, pad_en, ivl, amt
  g1.s1 = ((unsigned int)Kfull & 0xFFFFu) << 16;        // tensor_dim0 [15:0]
  g1.s2 = ((unsigned int)rows  & 0xFFFFu) << 16;        // tensor_dim0 hi=0 | tensor_dim1 lo
  g1.s3 = (32u << 16);                                  // tensor_dim1 hi=0 | tile_dim0=32
  g1.s4 = (unsigned int)rows;                           // tile_dim1=rows | tile_dim2=0
  g1.s5 = (unsigned int)Kfull;                          // tensor_dim0_stride[31:0]
  g1.s6 = 0u;                                           // stride0 hi | stride1 lo
  g1.s7 = 0u;                                           // stride1 hi
  asm volatile("tensor_load_to_lds %0, %1" :: "s"(g0), "s"(g1) : "memory");
}

__global__ __launch_bounds__(256)
void edge_transition(const float* __restrict__ z, const float* __restrict__ nb,
                     const _Float16* __restrict__ W1t, const float* __restrict__ b1,
                     const _Float16* __restrict__ W2t, const float* __restrict__ b2,
                     const _Float16* __restrict__ Wft, const float* __restrict__ bf,
                     const float* __restrict__ lng, const float* __restrict__ lnb,
                     float* __restrict__ eout)
{
  extern __shared__ __align__(16) char smem[];
  _Float16 (*eT)[EP]  = (_Float16(*)[EP])smem;                           // 16 x EP
  _Float16 (*hT)[EP]  = (_Float16(*)[EP])(smem + 16*EP*2);               // 16 x EP
  _Float16 (*Ws0)[WP] = (_Float16(*)[WP])(smem + 2*16*EP*2);             // 512 x WP
  _Float16 (*Ws1)[WP] = (_Float16(*)[WP])(smem + 2*16*EP*2 + 512*WP*2);  // 512 x WP
  float (*oT)[CZ]     = (float(*)[CZ])(smem + 2*16*EP*2 + 2*512*WP*2);   // 16 x 128
  _Float16 (*Wb[2])[WP] = { Ws0, Ws1 };

  const int tid = threadIdx.x;
  const int wave = tid >> 5, lane = tid & 31;
  const int ml = lane & 15, half = lane >> 4;
  const bool issuer = (__builtin_amdgcn_readfirstlane(wave) == 0);  // scalar branch
  const int i  = blockIdx.x / (NRES/16);
  const int j0 = (blockIdx.x % (NRES/16)) * 16;

  // phase 0: build e tile (f16)
  for (int li = tid; li < 16*HID; li += 256) {
    int r = li >> 9, c = li & 511;
    int j = j0 + r;
    float v;
    if (c < CZ)       v = z[((long)i*NRES + j)*CZ + c];
    else if (c < 320) v = nb[(long)i*192 + (c - 128)];
    else              v = nb[(long)j*192 + (c - 320)];
    eT[r][c] = (_Float16)v;
  }
  __syncthreads();

  // stages 1,2: h = relu(X @ W + b), 512 cols; wave owns 64 cols (4 tiles)
  for (int stage = 0; stage < 2; ++stage) {
    const _Float16* W = stage ? W2t : W1t;               // [512][512] n-major
    const float* bb   = stage ? b2 : b1;
    const _Float16 (*X)[EP] = stage ? (const _Float16(*)[EP])hT : (const _Float16(*)[EP])eT;
    v8f acc[4] = {};
    int cur = 0;
    if (issuer) tdm_slab(W, 512, HID, 0, Wb[0]);
    for (int k0 = 0; k0 < HID; k0 += 32) {
      if (issuer) TWAIT0();           // TDM slab landed (wave0's TENSORcnt)
      __syncthreads();                // release everyone
      if (issuer && (k0 + 32 < HID)) tdm_slab(W, 512, HID, k0 + 32, Wb[cur ^ 1]);
      v8h alo = *(const v8h*)&X[ml][k0 + half*8];
      v8h ahi = *(const v8h*)&X[ml][k0 + 16 + half*8];
      v16h af = cat8(alo, ahi);
      #pragma unroll
      for (int t4 = 0; t4 < 4; ++t4) {
        int ncol = wave*64 + t4*16 + ml;
        v8h blo = *(const v8h*)&Wb[cur][ncol][half*8];
        v8h bhi = *(const v8h*)&Wb[cur][ncol][16 + half*8];
        v16h bfr = cat8(blo, bhi);
        acc[t4] = __builtin_amdgcn_wmma_f32_16x16x32_f16(false, af, false, bfr, (short)0, acc[t4], false, false);
      }
      cur ^= 1;
    }
    __syncthreads();                  // all reads of X done before hT overwrite
    #pragma unroll
    for (int t4 = 0; t4 < 4; ++t4) {  // relu + bias -> hT
      int ncol = wave*64 + t4*16 + ml;
      float bv = bb[ncol];
      #pragma unroll
      for (int r = 0; r < 8; ++r) {
        float v = acc[t4][r] + bv;
        hT[half*8 + r][ncol] = (_Float16)(v > 0.f ? v : 0.f);
      }
    }
    __syncthreads();
  }

  // stage 3: (h + e) @ Wf + bf, 128 cols; wave -> one 16x16 tile
  // (weights staged with per-lane async loads: exercises ASYNCcnt path)
  {
    v8f acc = {};
    int cur = 0;
    async_slab(Wft, 128, HID, 0, Wb[0], tid);
    for (int k0 = 0; k0 < HID; k0 += 32) {
      AWAIT0();
      __syncthreads();
      if (k0 + 32 < HID) async_slab(Wft, 128, HID, k0 + 32, Wb[cur ^ 1], tid);
      v8h alo = *(const v8h*)&hT[ml][k0 + half*8];
      v8h ahi = *(const v8h*)&hT[ml][k0 + 16 + half*8];
      v8h elo = *(const v8h*)&eT[ml][k0 + half*8];
      v8h ehi = *(const v8h*)&eT[ml][k0 + 16 + half*8];
      v16h af = cat8(alo + elo, ahi + ehi);
      int ncol = wave*16 + ml;
      v8h blo = *(const v8h*)&Wb[cur][ncol][half*8];
      v8h bhi = *(const v8h*)&Wb[cur][ncol][16 + half*8];
      v16h bfr = cat8(blo, bhi);
      acc = __builtin_amdgcn_wmma_f32_16x16x32_f16(false, af, false, bfr, (short)0, acc, false, false);
      cur ^= 1;
    }
    __syncthreads();
    int ncol = wave*16 + ml;
    #pragma unroll
    for (int r = 0; r < 8; ++r) oT[half*8 + r][ncol] = acc[r] + bf[ncol];
  }
  __syncthreads();

  // phase 4: LayerNorm(128) per row; wave handles 2 rows, lane holds 4 cols
  #pragma unroll
  for (int rr = 0; rr < 2; ++rr) {
    int r = wave*2 + rr;
    float x[4], s = 0.f, s2 = 0.f;
    #pragma unroll
    for (int e = 0; e < 4; ++e) {
      x[e] = oT[r][lane + e*32];
      s += x[e]; s2 += x[e]*x[e];
    }
    for (int o = 16; o; o >>= 1) { s += __shfl_xor(s, o, 32); s2 += __shfl_xor(s2, o, 32); }
    float mean = s * (1.f/128.f);
    float var  = s2 * (1.f/128.f) - mean*mean;
    float rs = rsqrtf(var + 1e-5f);
    int j = j0 + r;
    float* orow = eout + ((long)i*NRES + j)*CZ;
    #pragma unroll
    for (int e = 0; e < 4; ++e) {
      int c = lane + e*32;
      orow[c] = (x[e] - mean)*rs*lng[c] + lnb[c];
    }
  }
}

// ---------------------------------------------------------------------------
static inline void launch_gemm(bool relu, const float* A, const float* W, const float* bias,
                               float* C, int M, int K, int Nc,
                               long ldA, long ldW, long ldC,
                               long sA, long sW, long sC, int batch, hipStream_t st)
{
  dim3 g((M + 63)/64, (Nc + 31)/32, batch);
  if (relu) gemm_wmma<true ><<<g, 256, 0, st>>>(A, W, bias, C, M, K, Nc, ldA, ldW, ldC, sA, sW, sC);
  else      gemm_wmma<false><<<g, 256, 0, st>>>(A, W, bias, C, M, K, Nc, ldA, ldW, ldC, sA, sW, sC);
}

extern "C" void kernel_launch(void* const* d_in, const int* in_sizes, int n_in,
                              void* d_out, int out_size, void* d_ws, size_t ws_size,
                              hipStream_t stream) {
  (void)in_sizes; (void)n_in; (void)out_size; (void)ws_size;
  const float* node  = (const float*)d_in[0];
  const float* edge  = (const float*)d_in[1];
  const float* rots  = (const float*)d_in[2];
  const float* trans = (const float*)d_in[3];
  const float* mask  = (const float*)d_in[4];
  const float *wq = (const float*)d_in[5],  *bq = (const float*)d_in[6];
  const float *wkv = (const float*)d_in[7], *bkv = (const float*)d_in[8];
  const float *wqp = (const float*)d_in[9], *bqp = (const float*)d_in[10];
  const float *wkvp = (const float*)d_in[11], *bkvp = (const float*)d_in[12];
  const float *wb = (const float*)d_in[13], *bb = (const float*)d_in[14];
  const float *head_w = (const float*)d_in[15];
  const float *wo = (const float*)d_in[16], *bo = (const float*)d_in[17];
  const float *ln_g = (const float*)d_in[18], *ln_b = (const float*)d_in[19];
  const float *nt_w1 = (const float*)d_in[20], *nt_b1 = (const float*)d_in[21];
  const float *nt_w2 = (const float*)d_in[22], *nt_b2 = (const float*)d_in[23];
  const float *nt_w3 = (const float*)d_in[24], *nt_b3 = (const float*)d_in[25];
  const float *nt_lng = (const float*)d_in[26], *nt_lnb = (const float*)d_in[27];
  const float *et_wi = (const float*)d_in[28], *et_bi = (const float*)d_in[29];
  const float *et_wt1 = (const float*)d_in[30], *et_bt1 = (const float*)d_in[31];
  const float *et_wt2 = (const float*)d_in[32], *et_bt2 = (const float*)d_in[33];
  const float *et_wf = (const float*)d_in[34], *et_bf = (const float*)d_in[35];
  const float *et_lng = (const float*)d_in[36], *et_lnb = (const float*)d_in[37];

  float* ws = (float*)d_ws;
  float* q_raw   = ws;
  float* kv_raw  = q_raw   + 384*256;
  float* qp_raw  = kv_raw  + 384*512;
  float* kvp_raw = qp_raw  + 384*384;
  float* qpts    = kvp_raw + 384*960;
  float* kvpts   = qpts    + 384*128*3;
  float* Qp      = kvpts   + 384*320*3;
  float* Kt      = Qp      + (long)16*384*64;
  float* Vc      = Kt      + (long)16*64*384;
  float* bz      = Vc      + (long)16*384*64;
  float* att     = bz      + (long)384*384*16;
  float* Outb    = att     + (long)16*384*384;
  float* Opair   = Outb    + (long)16*384*64;
  float* cat     = Opair   + (long)384*16*128;
  float* upd     = cat     + (long)384*3072;
  float* s1      = upd     + 384*384;
  float* t1      = s1      + 384*384;
  float* t2      = t1      + 384*384;
  float* t3      = t2      + 384*384;
  float* nb      = t3      + 384*384;
  _Float16* W1t  = (_Float16*)(nb + 384*192);      // 512x512 transposed f16
  _Float16* W2t  = W1t + 512*512;
  _Float16* Wft  = W2t + 512*512;                  // 128x512 transposed f16

  // one-time weight conversion for the fused edge kernel
  convert_transpose_f16<<<(512*512 + 255)/256, 256, 0, stream>>>(et_wt1, W1t, 512, 512);
  convert_transpose_f16<<<(512*512 + 255)/256, 256, 0, stream>>>(et_wt2, W2t, 512, 512);
  convert_transpose_f16<<<(512*128 + 255)/256, 256, 0, stream>>>(et_wf,  Wft, 512, 128);

  // node projections
  launch_gemm(false, node, wq,   bq,   q_raw,   384, 384, 256, 384, 256, 256, 0,0,0, 1, stream);
  launch_gemm(false, node, wkv,  bkv,  kv_raw,  384, 384, 512, 384, 512, 512, 0,0,0, 1, stream);
  launch_gemm(false, node, wqp,  bqp,  qp_raw,  384, 384, 384, 384, 384, 384, 0,0,0, 1, stream);
  launch_gemm(false, node, wkvp, bkvp, kvp_raw, 384, 384, 960, 384, 960, 960, 0,0,0, 1, stream);

  transform_points<<<(384*128 + 255)/256, 256, 0, stream>>>(qp_raw, rots, trans, qpts, 128);
  transform_points<<<(384*320 + 255)/256, 256, 0, stream>>>(kvp_raw, rots, trans, kvpts, 320);
  build_packs<<<(16*384 + 255)/256, 256, 0, stream>>>(q_raw, kv_raw, qpts, kvpts, head_w, Qp, Kt, Vc);

  // z-bias GEMM: [N*N,128] @ [128,16]
  launch_gemm(false, edge, wb, bb, bz, 384*384, 128, 16, 128, 16, 16, 0,0,0, 1, stream);

  // attention scores (batched per head): Qp[h] @ Kt[h]
  launch_gemm(false, Qp, Kt, nullptr, att, 384, 64, 384, 64, 384, 384,
              (long)384*64, (long)64*384, (long)384*384, 16, stream);
  softmax_rows<<<16*384, 128, 0, stream>>>(att, bz, mask);

  // o / o_pt (batched per head): a[h] @ Vcat[h]
  launch_gemm(false, att, Vc, nullptr, Outb, 384, 384, 64, 384, 64, 64,
              (long)384*384, (long)384*64, (long)384*64, 16, stream);
  // o_pair (batched per residue i): a[:,i,:] @ z[i]
  launch_gemm(false, att, edge, nullptr, Opair, 16, 384, 128,
              (long)384*384, 128, 128, 384, (long)384*128, (long)16*128, 384, stream);

  finalize_cat<<<384, 256, 0, stream>>>(Outb, Opair, rots, trans, cat);
  launch_gemm(false, cat, wo, bo, upd, 384, 3072, 384, 3072, 384, 384, 0,0,0, 1, stream);
  layernorm_row<0><<<384, 128, 0, stream>>>(node, upd, mask, ln_g, ln_b, s1, 384);

  // node transition MLP
  launch_gemm(true,  s1, nt_w1, nt_b1, t1, 384, 384, 384, 384, 384, 384, 0,0,0, 1, stream);
  launch_gemm(true,  t1, nt_w2, nt_b2, t2, 384, 384, 384, 384, 384, 384, 0,0,0, 1, stream);
  launch_gemm(false, t2, nt_w3, nt_b3, t3, 384, 384, 384, 384, 384, 384, 0,0,0, 1, stream);
  float* s_out = (float*)d_out;
  layernorm_row<1><<<384, 128, 0, stream>>>(s1, t3, mask, nt_lng, nt_lnb, s_out, 384);

  // edge transition (fused flagship kernel, TDM/async-staged weights)
  launch_gemm(false, s_out, et_wi, et_bi, nb, 384, 384, 192, 384, 192, 192, 0,0,0, 1, stream);
  float* e_out = s_out + 384*384;
  size_t smem = (size_t)2*16*EP*2 + (size_t)2*512*WP*2 + (size_t)16*CZ*4;  // ~120.5 KB < 320 KB/WGP
  edge_transition<<<384*(NRES/16), 256, smem, stream>>>(edge, nb, W1t, et_bt1,
                                                        W2t, et_bt2, Wft, et_bf,
                                                        et_lng, et_lnb, e_out);
}
